// SRULayer_5085241278584
// MI455X (gfx1250) — matrix operand order repaired and checked by reading, not compiled, not run on hardware
//
#include <hip/hip_runtime.h>
#include <hip/hip_bf16.h>

// ---------------------------------------------------------------------------
// SRU layer for MI455X (gfx1250, wave32, WMMA, async global->LDS).
//   L=2048, B=8, D=1024.  M = L*B = 16384 rows, GEMM N = 3*D = 3072, K = D.
// Pipeline: layernorm -> bf16 WMMA GEMM (+bias) -> fused sequential scan.
// ---------------------------------------------------------------------------

#define L_SEQ 2048
#define BATCH 8
#define DIM   1024
#define NDIM  (3 * DIM)          // 3072
#define MROWS (L_SEQ * BATCH)    // 16384

typedef __attribute__((ext_vector_type(16))) __bf16 bf16x16;
typedef __attribute__((ext_vector_type(8)))  float  v8f;

union FragU {
    bf16x16 v;
    uint4   u[2];
};

__device__ __forceinline__ unsigned short f2bf(float f) {
    unsigned int u = __float_as_uint(f);
    unsigned int r = u + 0x7fffu + ((u >> 16) & 1u);   // round-to-nearest-even
    return (unsigned short)(r >> 16);
}

// Async DMA of 16B/lane from global to LDS (GLOBAL_LOAD_ASYNC_TO_LDS_B128,
// tracked by ASYNCcnt).  lds_off is the LDS byte offset (generic shared
// pointer truncated to 32 bits == LDS aperture offset).
__device__ __forceinline__ void async_copy16(unsigned int lds_off,
                                             const void* gptr) {
    asm volatile("global_load_async_to_lds_b128 %0, %1, off"
                 :: "v"(lds_off), "v"(gptr)
                 : "memory");
}

__device__ __forceinline__ unsigned int lds_offset(const void* p) {
    return (unsigned int)(size_t)p;   // LDS aperture: addr[31:0] = LDS offset
}

// ---------------------------------------------------------------------------
// Kernel 1: LayerNorm over D, emit f32 x_norm (for epilogue blend) and bf16
// x_norm (WMMA A operand). One 256-thread block (8 wave32) per row.
// ---------------------------------------------------------------------------
__global__ __launch_bounds__(256) void sru_ln_kernel(
    const float* __restrict__ x, const float* __restrict__ gamma,
    const float* __restrict__ beta, float* __restrict__ xn,
    unsigned short* __restrict__ xnb)
{
    const int row = blockIdx.x;
    const int tid = threadIdx.x;

    const float4 v = ((const float4*)(x + (size_t)row * DIM))[tid];
    float s  = v.x + v.y + v.z + v.w;
    float s2 = v.x * v.x + v.y * v.y + v.z * v.z + v.w * v.w;

    // wave32 reduction
    #pragma unroll
    for (int off = 16; off > 0; off >>= 1) {
        s  += __shfl_down(s,  off, 32);
        s2 += __shfl_down(s2, off, 32);
    }
    __shared__ float red[2][8];
    const int wid = tid >> 5, lane = tid & 31;
    if (lane == 0) { red[0][wid] = s; red[1][wid] = s2; }
    __syncthreads();
    float ts = 0.f, ts2 = 0.f;
    #pragma unroll
    for (int i = 0; i < 8; ++i) { ts += red[0][i]; ts2 += red[1][i]; }

    const float mean = ts * (1.0f / DIM);
    const float var  = ts2 * (1.0f / DIM) - mean * mean;
    const float inv  = rsqrtf(var + 1e-5f);

    const float4 g = ((const float4*)gamma)[tid];
    const float4 b = ((const float4*)beta)[tid];
    float4 y;
    y.x = (v.x - mean) * inv * g.x + b.x;
    y.y = (v.y - mean) * inv * g.y + b.y;
    y.z = (v.z - mean) * inv * g.z + b.z;
    y.w = (v.w - mean) * inv * g.w + b.w;

    ((float4*)(xn + (size_t)row * DIM))[tid] = y;
    ushort4 o;
    o.x = f2bf(y.x); o.y = f2bf(y.y); o.z = f2bf(y.z); o.w = f2bf(y.w);
    ((ushort4*)(xnb + (size_t)row * DIM))[tid] = o;
}

// ---------------------------------------------------------------------------
// Kernel 2: W (fp32 [3072,1024], K-contiguous) -> bf16 for WMMA B operand.
// ---------------------------------------------------------------------------
__global__ __launch_bounds__(256) void sru_wconv_kernel(
    const float* __restrict__ W, unsigned short* __restrict__ Wb)
{
    const int i = blockIdx.x * 256 + threadIdx.x;   // one float4 per thread
    const float4 v = ((const float4*)W)[i];
    ushort4 o;
    o.x = f2bf(v.x); o.y = f2bf(v.y); o.z = f2bf(v.z); o.w = f2bf(v.w);
    ((ushort4*)Wb)[i] = o;
}

// ---------------------------------------------------------------------------
// Kernel 3: GEMM  ufr[m,n] = sum_k A[m,k]*W[n,k] + bias[n]
//   A: bf16 [16384,1024]   W: bf16 [3072,1024]   ufr: f32 [16384,3072]
// 128x128 block tile, 8 waves in 4x2, each wave 32x64 (2x4 WMMA 16x16 accs),
// K-step 32 (one v_wmma_f32_16x16x32_bf16 depth).
// Double-buffered LDS tiles filled by GLOBAL_LOAD_ASYNC_TO_LDS_B128:
// stage k+1 is in flight (ASYNCcnt) while stage k feeds the WMMAs.
// K-loop unrolled by 2 (+ peeled tail) so buffer indices are compile-time:
// no cndmask address selects, no branches in the steady state.
// LDS row stride 40 bf16 (=20 dwords): keeps rows 16B-aligned for
// ds_load_b128 and makes the 16-lane fragment reads bank-conflict-free.
// ---------------------------------------------------------------------------
#define LDS_STRIDE 40
#define NK (DIM / 32)

__device__ __forceinline__ bf16x16 load_a_frag(const unsigned short* lds,
                                               int row_base, int lane)
{
    const int half = lane >> 4;
    const int m    = row_base + (lane & 15);
    // CDNA5 16-bit A 16x32 layout: lanes 0-15 hold K 0-7 / 16-23,
    // lanes 16-31 hold K 8-15 / 24-31 (2 bf16 per VGPR, lo = even K).
    const int k0 = half ? 8 : 0;
    const int k1 = half ? 24 : 16;
    FragU fr;
    fr.u[0] = *(const uint4*)(lds + (size_t)m * LDS_STRIDE + k0);
    fr.u[1] = *(const uint4*)(lds + (size_t)m * LDS_STRIDE + k1);
    return fr.v;
}

__device__ __forceinline__ bf16x16 load_b_frag(const unsigned short* lds,
                                               int col_base, int lane)
{
    const int half = lane >> 4;
    const int n    = col_base + (lane & 15);
    // CDNA5 16-bit B 32x16 layout: lanes 0-15 (N=lane) hold K 0-7 / 16-23,
    // lanes 16-31 (N=lane-16) hold K 8-15 / 24-31.
    const int kb = half << 3;
    FragU fr;
    fr.u[0] = *(const uint4*)(lds + (size_t)n * LDS_STRIDE + kb);
    fr.u[1] = *(const uint4*)(lds + (size_t)n * LDS_STRIDE + kb + 16);
    return fr.v;
}

__global__ __launch_bounds__(256) void sru_gemm_kernel(
    const unsigned short* __restrict__ A, const unsigned short* __restrict__ Bw,
    const float* __restrict__ bias, float* __restrict__ C)
{
    __shared__ unsigned short lds_a[2][128 * LDS_STRIDE];
    __shared__ unsigned short lds_b[2][128 * LDS_STRIDE];   // 4 x 10 KB total

    const int tid  = threadIdx.x;
    const int lane = tid & 31;
    const int wid  = tid >> 5;
    const int wm   = wid & 3;        // wave row  (0..3) -> 32 rows each
    const int wn   = wid >> 2;       // wave col  (0..1) -> 64 cols each
    const int tm   = blockIdx.x * 128;
    const int tn   = blockIdx.y * 128;

    // staging map: 512 x 16B chunks per tile; each thread owns chunks at
    // rows r0 and r0+64, 16B-chunk ch0 within the 64B row.
    const int r0  = tid >> 2;
    const int ch0 = tid & 3;

    const unsigned short* ga0 = A  + (size_t)(tm + r0)      * DIM + ch0 * 8;
    const unsigned short* ga1 = A  + (size_t)(tm + r0 + 64) * DIM + ch0 * 8;
    const unsigned short* gb0 = Bw + (size_t)(tn + r0)      * DIM + ch0 * 8;
    const unsigned short* gb1 = Bw + (size_t)(tn + r0 + 64) * DIM + ch0 * 8;

    unsigned int lA0[2], lA1[2], lB0[2], lB1[2];
    #pragma unroll
    for (int p = 0; p < 2; ++p) {
        lA0[p] = lds_offset(&lds_a[p][(size_t)r0        * LDS_STRIDE + ch0 * 8]);
        lA1[p] = lds_offset(&lds_a[p][(size_t)(r0 + 64) * LDS_STRIDE + ch0 * 8]);
        lB0[p] = lds_offset(&lds_b[p][(size_t)r0        * LDS_STRIDE + ch0 * 8]);
        lB1[p] = lds_offset(&lds_b[p][(size_t)(r0 + 64) * LDS_STRIDE + ch0 * 8]);
    }

    v8f acc[2][4] = {};

    // issue one K-stage (4 async b128 ops per thread) into buffer p
    auto stage = [&](int s, int p) {
        const size_t ko = (size_t)s * 32;   // bf16 elements
        async_copy16(lA0[p], ga0 + ko);
        async_copy16(lA1[p], ga1 + ko);
        async_copy16(lB0[p], gb0 + ko);
        async_copy16(lB1[p], gb1 + ko);
    };
    // consume buffer p: 6 fragment ds_load_b128 pairs + 8 WMMAs
    auto compute = [&](int p) {
        bf16x16 af[2];
        af[0] = load_a_frag(lds_a[p], wm * 32, lane);
        af[1] = load_a_frag(lds_a[p], wm * 32 + 16, lane);
        bf16x16 bf[4];
        #pragma unroll
        for (int j = 0; j < 4; ++j)
            bf[j] = load_b_frag(lds_b[p], wn * 64 + j * 16, lane);
        #pragma unroll
        for (int i = 0; i < 2; ++i)
            #pragma unroll
            for (int j = 0; j < 4; ++j)
                acc[i][j] = __builtin_amdgcn_wmma_f32_16x16x32_bf16(
                    false, af[i], false, bf[j], (short)0, acc[i][j],
                    false, false);
    };

    stage(0, 0);

    #pragma unroll 1
    for (int kt = 0; kt < NK - 2; kt += 2) {
        stage(kt + 1, 1);
        asm volatile("s_wait_asynccnt 0x4" ::: "memory");  // stage kt resident
        __syncthreads();
        compute(0);
        __syncthreads();   // buffer 0 reads done before re-targeting it

        stage(kt + 2, 0);
        asm volatile("s_wait_asynccnt 0x4" ::: "memory");  // stage kt+1 resident
        __syncthreads();
        compute(1);
        __syncthreads();   // buffer 1 reads done before re-targeting it
    }
    // peeled tail: stages NK-2 (buf0, in flight) and NK-1 (buf1)
    stage(NK - 1, 1);
    asm volatile("s_wait_asynccnt 0x4" ::: "memory");
    __syncthreads();
    compute(0);
    __syncthreads();
    asm volatile("s_wait_asynccnt 0x0" ::: "memory");
    __syncthreads();
    compute(1);

    // epilogue: C/D layout — VGPR v, lanes 0-15: M=v, N=lane;
    //                        lanes 16-31: M=8+v, N=lane-16.
    const int half = lane >> 4;
    const int nl   = lane & 15;
    #pragma unroll
    for (int i = 0; i < 2; ++i) {
        #pragma unroll
        for (int j = 0; j < 4; ++j) {
            const int gn = tn + wn * 64 + j * 16 + nl;
            const float bv = bias[gn];
            const int rbase = tm + wm * 32 + i * 16 + half * 8;
            float* cp = C + (size_t)rbase * NDIM + gn;
            #pragma unroll
            for (int v = 0; v < 8; ++v)
                cp[(size_t)v * NDIM] = acc[i][j][v] + bv;
        }
    }
}

// ---------------------------------------------------------------------------
// Kernel 4: fused recurrence + epilogue. One thread per (b,d) = 8192 chains.
//   c_t = f*c + (1-f)*u ; h = r*tanh(c_t) + (1-r)*xn ; out = x + h
// Streams 384 MB; global_prefetch_b8 issued ahead to hide latency.
// ---------------------------------------------------------------------------
__global__ __launch_bounds__(256) void sru_scan_kernel(
    const float* __restrict__ x, const float* __restrict__ c0,
    const float* __restrict__ ufr, const float* __restrict__ xn,
    float* __restrict__ out, float* __restrict__ lastc)
{
    const int idx = blockIdx.x * 256 + threadIdx.x;   // 0..8191
    const int b = idx >> 10;
    const int d = idx & (DIM - 1);

    float c = c0[idx];
    const float* pu = ufr + (size_t)b * NDIM + d;
    const float* pn = xn  + (size_t)b * DIM + d;
    const float* px = x   + (size_t)b * DIM + d;
    float*       po = out + (size_t)b * DIM + d;
    const size_t su = (size_t)BATCH * NDIM;   // l-stride in ufr
    const size_t sx = (size_t)BATCH * DIM;    // l-stride in x/xn/out

    #pragma unroll 1
    for (int l = 0; l < L_SEQ; ++l) {
        __builtin_prefetch(pu + 4 * su, 0, 0);
        __builtin_prefetch(pu + 4 * su + DIM, 0, 0);
        __builtin_prefetch(pu + 4 * su + 2 * DIM, 0, 0);
        __builtin_prefetch(pn + 4 * sx, 0, 0);
        __builtin_prefetch(px + 4 * sx, 0, 0);

        const float u  = pu[0];
        const float fl = pu[DIM];
        const float rl = pu[2 * DIM];
        const float nv = pn[0];
        const float xv = px[0];

        const float f = 1.0f / (1.0f + __expf(-fl));
        const float r = 1.0f / (1.0f + __expf(-rl));
        c = fmaf(f, c - u, u);                       // f*c + (1-f)*u
        const float h = fmaf(r, tanhf(c) - nv, nv);  // r*tanh(c) + (1-r)*xn
        po[0] = xv + h;

        pu += su; pn += sx; px += sx; po += sx;
    }
    lastc[idx] = c;
}

// ---------------------------------------------------------------------------
// Host-side launch.  d_in: x, c, W, b, gamma, beta (fp32).
// d_out: out [L,B,D] then last_c [B,D].  Workspace layout (bytes):
//   [0, 64MB)        x_norm  f32
//   [64MB, 256MB)    ufr     f32
//   [256MB, 288MB)   x_norm  bf16
//   [288MB, 294MB)   W       bf16
// ---------------------------------------------------------------------------
extern "C" void kernel_launch(void* const* d_in, const int* in_sizes, int n_in,
                              void* d_out, int out_size, void* d_ws, size_t ws_size,
                              hipStream_t stream) {
    const float* x     = (const float*)d_in[0];
    const float* c0    = (const float*)d_in[1];
    const float* W     = (const float*)d_in[2];
    const float* bias  = (const float*)d_in[3];
    const float* gamma = (const float*)d_in[4];
    const float* beta  = (const float*)d_in[5];

    char* ws = (char*)d_ws;
    float*          xn  = (float*)ws;
    float*          ufr = (float*)(ws + (size_t)67108864);
    unsigned short* xnb = (unsigned short*)(ws + (size_t)67108864 + 201326592);
    unsigned short* wb  = (unsigned short*)(ws + (size_t)67108864 + 201326592 + 33554432);

    float* out   = (float*)d_out;
    float* lastc = out + (size_t)MROWS * DIM;

    sru_ln_kernel<<<dim3(MROWS), dim3(256), 0, stream>>>(x, gamma, beta, xn, xnb);
    sru_wconv_kernel<<<dim3((NDIM * DIM) / (256 * 4)), dim3(256), 0, stream>>>(W, wb);
    sru_gemm_kernel<<<dim3(MROWS / 128, NDIM / 128), dim3(256), 0, stream>>>(xnb, wb, bias, ufr);
    sru_scan_kernel<<<dim3((BATCH * DIM) / 256), dim3(256), 0, stream>>>(x, c0, ufr, xn, out, lastc);
}
